// MultiHeadAttention_87926570484461
// MI455X (gfx1250) — matrix operand order repaired and checked by reading, not compiled, not run on hardware
//
#include <hip/hip_runtime.h>
#include <stdint.h>

typedef __attribute__((ext_vector_type(16))) _Float16 v16h;
typedef __attribute__((ext_vector_type(8)))  _Float16 v8h;
typedef __attribute__((ext_vector_type(8)))  float    v8f;
typedef int i32x4 __attribute__((vector_size(4 * sizeof(int))));

namespace {
constexpr int BB = 2;
constexpr int TT = 2048;
constexpr int DD = 1024;
constexpr int HH = 16;
constexpr int DK = 64;
constexpr int PPITCH = 136;   // 128 + 8 halves pad: bank-conflict-free ds reads
}

#define AS1 __attribute__((address_space(1)))
#define AS3 __attribute__((address_space(3)))

#if defined(__AMDGCN__) && __has_builtin(__builtin_amdgcn_global_load_async_to_lds_b128)
#define HAS_ASYNC 1
#else
#define HAS_ASYNC 0
#endif

// ---- CDNA5 async global->LDS staging (ASYNCcnt path), with safe fallback ----
__device__ __forceinline__ void stage_b128(_Float16* ldst, const _Float16* gsrc) {
#if HAS_ASYNC
    __builtin_amdgcn_global_load_async_to_lds_b128(
        (AS1 i32x4*)(uintptr_t)gsrc, (AS3 i32x4*)(uintptr_t)ldst, 0, 0);
#else
    *(v8h*)ldst = *(const v8h*)gsrc;
#endif
}

__device__ __forceinline__ void stage_wait() {
#if HAS_ASYNC
#if __has_builtin(__builtin_amdgcn_s_wait_asynccnt)
    __builtin_amdgcn_s_wait_asynccnt(0);
#else
    asm volatile("s_wait_asynccnt 0x0" ::: "memory");
#endif
#endif
    __syncthreads();
}

__device__ __forceinline__ v8f wmma_f16(v16h a, v16h b, v8f c) {
    return __builtin_amdgcn_wmma_f32_16x16x32_f16(
        false, a, false, b, (short)0, c, false, false);
}

// ---- fragment loaders -------------------------------------------------------
// A fragment 16x32: p = row(m0 + lane%16) + k_base + (lane/16)*8
__device__ __forceinline__ v16h frag_a_f16(const _Float16* __restrict__ p) {
    v8h x = *(const v8h*)p;
    v8h y = *(const v8h*)(p + 16);
    v16h f;
#pragma unroll
    for (int i = 0; i < 8; ++i) { f[i] = x[i]; f[i + 8] = y[i]; }
    return f;
}

__device__ __forceinline__ v16h frag_a_f32(const float* __restrict__ p) {
    v16h f;
#pragma unroll
    for (int i = 0; i < 8; ++i) {
        f[i]     = (_Float16)p[i];
        f[i + 8] = (_Float16)p[i + 16];
    }
    return f;
}

// B fragment 32x16: lane holds K {0..7,16..23} (+8 for upper half) of column n = lane%16,
// p = S_row(n0 + lane%16) + k_base + (lane/16)*8. Two v8h: 16B alignment OK
// for both global and padded-LDS sources.
__device__ __forceinline__ v16h frag_b(const _Float16* __restrict__ p) {
    v8h x = *(const v8h*)p;
    v8h y = *(const v8h*)(p + 16);
    v16h f;
#pragma unroll
    for (int i = 0; i < 8; ++i) { f[i] = x[i]; f[i + 8] = y[i]; }
    return f;
}

// ---- f32 -> f16 convert (weights + input activations, one time) -------------
__global__ __launch_bounds__(256) void k_cvt(
    const float* __restrict__ s, _Float16* __restrict__ d, int n)
{
    int i = (blockIdx.x * 256 + threadIdx.x) * 8;
    if (i + 8 <= n) {
        v8h o;
#pragma unroll
        for (int j = 0; j < 8; ++j) o[j] = (_Float16)s[i + j];
        *(v8h*)(d + i) = o;
    }
}

// ---- shared GEMM core: 8 waves x (32x64) tiles, B panel staged via LDS ------
// A0/A1: rows (m0+lr) and (m0+16+lr) of X (f16, ld = K), pre-offset by lh*8.
// Wn0: W + n0*K (row-major [N,K] f16). Accumulates acc[0..3] (rows m0..m0+15)
// and acc[4..7] (rows m0+16..m0+31) over columns n0 + j*16 + lane%16.
__device__ __forceinline__ void gemm_panel(
    const _Float16* __restrict__ A0, const _Float16* __restrict__ A1,
    const _Float16* __restrict__ Wn0, _Float16* lds, int K,
    int tid, int lr, int lh, v8f acc[8])
{
    for (int kc = 0; kc < K; kc += 128) {
        // stage 64 x 128 f16 panel: 1024 x 16B chunks, 4 per thread
#pragma unroll
        for (int i = 0; i < 4; ++i) {
            int c = tid + 256 * i;
            int row = c >> 4, kp = c & 15;
            stage_b128(lds + row * PPITCH + kp * 8,
                       Wn0 + (size_t)row * K + kc + kp * 8);
        }
        stage_wait();
#pragma unroll
        for (int kk = 0; kk < 128; kk += 32) {
            v16h a0 = frag_a_f16(A0 + kc + kk);
            v16h a1 = frag_a_f16(A1 + kc + kk);
#pragma unroll
            for (int j = 0; j < 4; ++j) {
                v16h b = frag_b(lds + (j * 16 + lr) * PPITCH + kk + lh * 8);
                acc[j]     = wmma_f16(a0, b, acc[j]);
                acc[4 + j] = wmma_f16(a1, b, acc[4 + j]);
            }
        }
        __syncthreads();
    }
}

// ---- kernel: Q/K projection, Y[M=B*T, DD] f16 = X f16 @ W^T f16 -------------
__global__ __launch_bounds__(256) void k_proj_qk(
    const _Float16* __restrict__ X, const _Float16* __restrict__ Wh,
    _Float16* __restrict__ Y)
{
    __shared__ _Float16 panel[64 * PPITCH];
    const int tid = threadIdx.x;
    const int lane = tid & 31, warp = tid >> 5;
    const int lr = lane & 15, lh = lane >> 4;
    const int n0 = (blockIdx.x & 15) << 6;               // 16 n-panels
    const int m0 = ((blockIdx.x >> 4) << 8) + warp * 32; // 16 m-blocks of 256

    v8f acc[8] = {};
    gemm_panel(X + (size_t)(m0 + lr) * DD + lh * 8,
               X + (size_t)(m0 + 16 + lr) * DD + lh * 8,
               Wh + (size_t)n0 * DD, panel, DD, tid, lr, lh, acc);

#pragma unroll
    for (int mi = 0; mi < 2; ++mi)
#pragma unroll
        for (int j = 0; j < 4; ++j)
#pragma unroll
            for (int v = 0; v < 8; ++v)
                Y[(size_t)(m0 + mi * 16 + lh * 8 + v) * DD + n0 + j * 16 + lr] =
                    (_Float16)acc[mi * 4 + j][v];
}

// ---- kernel: V projection stored transposed per head: Vt[B][H][DK][T] -------
__global__ __launch_bounds__(256) void k_proj_vt(
    const _Float16* __restrict__ X, const _Float16* __restrict__ Wh,
    _Float16* __restrict__ Vt)
{
    __shared__ _Float16 panel[64 * PPITCH];
    const int tid = threadIdx.x;
    const int lane = tid & 31, warp = tid >> 5;
    const int lr = lane & 15, lh = lane >> 4;
    const int h  = blockIdx.x & 15;                      // 64-wide panel == head
    const int m0 = ((blockIdx.x >> 4) << 8) + warp * 32;
    const int b  = m0 / TT, t0 = m0 % TT;

    v8f acc[8] = {};
    gemm_panel(X + (size_t)(m0 + lr) * DD + lh * 8,
               X + (size_t)(m0 + 16 + lr) * DD + lh * 8,
               Wh + (size_t)(h * 64) * DD, panel, DD, tid, lr, lh, acc);

    _Float16* hb = Vt + (size_t)(b * HH + h) * DK * TT;
#pragma unroll
    for (int mi = 0; mi < 2; ++mi)
#pragma unroll
        for (int j = 0; j < 4; ++j) {
            v8h o;
#pragma unroll
            for (int v = 0; v < 8; ++v) o[v] = (_Float16)acc[mi * 4 + j][v];
            *(v8h*)(hb + (size_t)(j * 16 + lr) * TT + t0 + mi * 16 + lh * 8) = o;
        }
}

// ---- kernel: scores = Q @ K^T * 0.125, f32 to weights region ----------------
__global__ __launch_bounds__(256) void k_scores(
    const _Float16* __restrict__ Q, const _Float16* __restrict__ Km,
    float* __restrict__ Wt)
{
    const int lane = threadIdx.x & 31, warp = threadIdx.x >> 5;
    const int lr = lane & 15, lh = lane >> 4;
    const int bh = blockIdx.y;
    const int b = bh / HH, h = bh % HH;
    const int wid = blockIdx.x * 8 + warp;
    const int nt = TT >> 6;
    const int m0 = (wid / nt) << 5;
    const int n0 = (wid % nt) << 6;

    const _Float16* qb = Q  + (size_t)b * TT * DD + h * DK;
    const _Float16* kb = Km + (size_t)b * TT * DD + h * DK;

    v8f acc[8] = {};
#pragma unroll
    for (int k = 0; k < DK; k += 32) {
        v16h a0 = frag_a_f16(qb + (size_t)(m0 + lr) * DD + k + lh * 8);
        v16h a1 = frag_a_f16(qb + (size_t)(m0 + 16 + lr) * DD + k + lh * 8);
#pragma unroll
        for (int j = 0; j < 4; ++j) {
            v16h bf = frag_b(kb + (size_t)(n0 + j * 16 + lr) * DD + k + lh * 8);
            acc[j]     = wmma_f16(a0, bf, acc[j]);
            acc[4 + j] = wmma_f16(a1, bf, acc[4 + j]);
        }
    }
    float* wb = Wt + ((size_t)bh * TT + m0) * TT;
    const float scale = 0.125f;
#pragma unroll
    for (int mi = 0; mi < 2; ++mi)
#pragma unroll
        for (int j = 0; j < 4; ++j)
#pragma unroll
            for (int v = 0; v < 8; ++v)
                wb[(size_t)(mi * 16 + lh * 8 + v) * TT + n0 + j * 16 + lr] =
                    acc[mi * 4 + j][v] * scale;
}

// ---- kernel: row softmax in place ------------------------------------------
__global__ __launch_bounds__(256) void k_softmax(float* __restrict__ Wt)
{
    float* row = Wt + (size_t)blockIdx.x * TT;
    const int tid = threadIdx.x;
    __shared__ float red[8];
    __shared__ float bc;

    float vals[8];
    float mx = -3.4e38f;
#pragma unroll
    for (int i = 0; i < 8; ++i) { vals[i] = row[tid + i * 256]; mx = fmaxf(mx, vals[i]); }
#pragma unroll
    for (int o = 16; o >= 1; o >>= 1) mx = fmaxf(mx, __shfl_xor(mx, o, 32));
    if ((tid & 31) == 0) red[tid >> 5] = mx;
    __syncthreads();
    if (tid == 0) {
        float m = red[0];
        for (int i = 1; i < 8; ++i) m = fmaxf(m, red[i]);
        bc = m;
    }
    __syncthreads();
    mx = bc;

    float s = 0.f;
#pragma unroll
    for (int i = 0; i < 8; ++i) { vals[i] = __expf(vals[i] - mx); s += vals[i]; }
#pragma unroll
    for (int o = 16; o >= 1; o >>= 1) s += __shfl_xor(s, o, 32);
    __syncthreads();
    if ((tid & 31) == 0) red[tid >> 5] = s;
    __syncthreads();
    if (tid == 0) {
        float t = 0.f;
        for (int i = 0; i < 8; ++i) t += red[i];
        bc = t;
    }
    __syncthreads();
    const float inv = 1.0f / bc;
#pragma unroll
    for (int i = 0; i < 8; ++i) row[tid + i * 256] = vals[i] * inv;
}

// ---- kernel: attn = P @ V (P f32->f16 on load, Vt panel staged in LDS) ------
__global__ __launch_bounds__(256) void k_pv(
    const float* __restrict__ P, const _Float16* __restrict__ Vt,
    _Float16* __restrict__ A)
{
    __shared__ _Float16 panel[64 * PPITCH];
    const int tid = threadIdx.x;
    const int lane = tid & 31, warp = tid >> 5;
    const int lr = lane & 15, lh = lane >> 4;
    const int bh = blockIdx.y;
    const int b = bh / HH, h = bh % HH;
    const int m0 = blockIdx.x * 256 + warp * 32;

    const float*    pb = P  + (size_t)bh * TT * TT;
    const _Float16* vb = Vt + (size_t)bh * DK * TT;

    v8f acc[8] = {};
    const float* a0p = pb + (size_t)(m0 + lr) * TT + lh * 8;
    const float* a1p = pb + (size_t)(m0 + 16 + lr) * TT + lh * 8;

    for (int kc = 0; kc < TT; kc += 128) {
#pragma unroll
        for (int i = 0; i < 4; ++i) {
            int c = tid + 256 * i;
            int row = c >> 4, kp = c & 15;
            stage_b128(panel + row * PPITCH + kp * 8,
                       vb + (size_t)row * TT + kc + kp * 8);
        }
        stage_wait();
#pragma unroll
        for (int kk = 0; kk < 128; kk += 32) {
            v16h a0 = frag_a_f32(a0p + kc + kk);
            v16h a1 = frag_a_f32(a1p + kc + kk);
#pragma unroll
            for (int j = 0; j < 4; ++j) {
                v16h bf = frag_b(panel + (j * 16 + lr) * PPITCH + kk + lh * 8);
                acc[j]     = wmma_f16(a0, bf, acc[j]);
                acc[4 + j] = wmma_f16(a1, bf, acc[4 + j]);
            }
        }
        __syncthreads();
    }
    _Float16* ab = A + (size_t)(b * TT + m0) * DD + h * DK;
#pragma unroll
    for (int mi = 0; mi < 2; ++mi)
#pragma unroll
        for (int j = 0; j < 4; ++j)
#pragma unroll
            for (int v = 0; v < 8; ++v)
                ab[(size_t)(mi * 16 + lh * 8 + v) * DD + j * 16 + lr] =
                    (_Float16)acc[mi * 4 + j][v];
}

// ---- kernel: out f32 = attn f16 @ W_o^T f16 ---------------------------------
__global__ __launch_bounds__(256) void k_oproj(
    const _Float16* __restrict__ X, const _Float16* __restrict__ Wh,
    float* __restrict__ Y)
{
    __shared__ _Float16 panel[64 * PPITCH];
    const int tid = threadIdx.x;
    const int lane = tid & 31, warp = tid >> 5;
    const int lr = lane & 15, lh = lane >> 4;
    const int n0 = (blockIdx.x & 15) << 6;
    const int m0 = ((blockIdx.x >> 4) << 8) + warp * 32;

    v8f acc[8] = {};
    gemm_panel(X + (size_t)(m0 + lr) * DD + lh * 8,
               X + (size_t)(m0 + 16 + lr) * DD + lh * 8,
               Wh + (size_t)n0 * DD, panel, DD, tid, lr, lh, acc);

#pragma unroll
    for (int mi = 0; mi < 2; ++mi)
#pragma unroll
        for (int j = 0; j < 4; ++j)
#pragma unroll
            for (int v = 0; v < 8; ++v)
                Y[(size_t)(m0 + mi * 16 + lh * 8 + v) * DD + n0 + j * 16 + lr] =
                    acc[mi * 4 + j][v];
}

// ---- host launcher ----------------------------------------------------------
extern "C" void kernel_launch(void* const* d_in, const int* in_sizes, int n_in,
                              void* d_out, int out_size, void* d_ws, size_t ws_size,
                              hipStream_t stream)
{
    (void)in_sizes; (void)n_in; (void)out_size; (void)ws_size;
    const float* q  = (const float*)d_in[0];
    const float* k  = (const float*)d_in[1];
    const float* v  = (const float*)d_in[2];
    const float* Wq = (const float*)d_in[3];
    const float* Wk = (const float*)d_in[4];
    const float* Wv = (const float*)d_in[5];
    const float* Wo = (const float*)d_in[6];

    float* out     = (float*)d_out;                 // [B,T,D]
    float* weights = out + (size_t)BB * TT * DD;    // [B,H,T,T]

    constexpr size_t NTD = (size_t)BB * TT * DD;    // 4M
    constexpr size_t ND2 = (size_t)DD * DD;         // 1M
    _Float16* ws  = (_Float16*)d_ws;
    _Float16* qx  = ws;                 // inputs converted to f16
    _Float16* kx  = qx + NTD;
    _Float16* vx  = kx + NTD;
    _Float16* wqh = vx + NTD;           // weights converted to f16
    _Float16* wkh = wqh + ND2;
    _Float16* wvh = wkh + ND2;
    _Float16* woh = wvh + ND2;
    _Float16* qw  = woh + ND2;          // projected Q [B,T,D]
    _Float16* kw  = qw + NTD;           // projected K [B,T,D]
    _Float16* vt  = kw + NTD;           // projected V^T [B,H,DK,T]
    _Float16* at  = vt + NTD;           // attn [B,T,D]

    const dim3 blk(256);

    // one-time f32 -> f16 conversion
    k_cvt<<<dim3((int)(NTD / 2048)), blk, 0, stream>>>(q,  qx,  (int)NTD);
    k_cvt<<<dim3((int)(NTD / 2048)), blk, 0, stream>>>(k,  kx,  (int)NTD);
    k_cvt<<<dim3((int)(NTD / 2048)), blk, 0, stream>>>(v,  vx,  (int)NTD);
    k_cvt<<<dim3((int)(ND2 / 2048)), blk, 0, stream>>>(Wq, wqh, (int)ND2);
    k_cvt<<<dim3((int)(ND2 / 2048)), blk, 0, stream>>>(Wk, wkh, (int)ND2);
    k_cvt<<<dim3((int)(ND2 / 2048)), blk, 0, stream>>>(Wv, wvh, (int)ND2);
    k_cvt<<<dim3((int)(ND2 / 2048)), blk, 0, stream>>>(Wo, woh, (int)ND2);

    // projections: 16 m-blocks(256 rows) x 16 n-panels(64)
    k_proj_qk<<<dim3(256), blk, 0, stream>>>(qx, wqh, qw);
    k_proj_qk<<<dim3(256), blk, 0, stream>>>(kx, wkh, kw);
    k_proj_vt<<<dim3(256), blk, 0, stream>>>(vx, wvh, vt);

    // scores + softmax (weights live in d_out)
    k_scores<<<dim3((TT / 32) * (TT / 64) / 8, BB * HH), blk, 0, stream>>>(qw, kw, weights);
    k_softmax<<<dim3(BB * HH * TT), blk, 0, stream>>>(weights);

    // attn = P @ V, then output projection
    k_pv<<<dim3(TT / 256, BB * HH), blk, 0, stream>>>(weights, vt, at);
    k_oproj<<<dim3(256), blk, 0, stream>>>(at, woh, out);
}
